// LinearDeepSeekV3_22153441312854
// MI455X (gfx1250) — 8-bit matrix operand order repaired and checked by reading, compiled, not run on hardware
//
#include <hip/hip_runtime.h>


typedef int   v16i __attribute__((ext_vector_type(16)));
typedef float v8f  __attribute__((ext_vector_type(8)));

#define K_DIM       7168
#define N_DIM       256
#define KB          56
#define BLK         128
#define ROWS_WG     64
#define XQ_STRIDE   136
#define STAGE_BYTES (256 * 128)

__device__ __forceinline__ void async_stage_x(unsigned lds_off, const float* gp)
{
    asm volatile("global_load_async_to_lds_b128 %0, %1, off\n\t"
                 "global_load_async_to_lds_b128 %0, %1, off offset:16\n\t"
                 "global_load_async_to_lds_b128 %0, %1, off offset:32\n\t"
                 "global_load_async_to_lds_b128 %0, %1, off offset:48\n\t"
                 "global_load_async_to_lds_b128 %0, %1, off offset:64\n\t"
                 "global_load_async_to_lds_b128 %0, %1, off offset:80\n\t"
                 "global_load_async_to_lds_b128 %0, %1, off offset:96\n\t"
                 "global_load_async_to_lds_b128 %0, %1, off offset:112"
                 :: "v"(lds_off), "v"(gp) : "memory");
}

__global__ __launch_bounds__(256)
void fp8_gemm_dsv3_kernel(const float* __restrict__ x,
                          const unsigned char* __restrict__ w,
                          const float* __restrict__ wscale,
                          float* __restrict__ out,
                          int M)
{
    __shared__ __align__(16) unsigned char lds_stage[2 * STAGE_BYTES];
    __shared__ __align__(16) unsigned char lds_xq[ROWS_WG * XQ_STRIDE];
    __shared__ float lds_as[ROWS_WG];

    const int tid    = threadIdx.x;
    const int lane   = tid & 31;
    const int wave   = tid >> 5;
    const int wave_m = wave & 3;
    const int wave_n = wave >> 2;
    const int half   = lane >> 4;
    const int l16    = lane & 15;

    const int row0 = blockIdx.x * ROWS_WG;

    const int qrow = tid >> 2;
    const int qseg = tid & 3;
    const float* xrow = x + (size_t)(row0 + qrow) * K_DIM + qseg * 32;

    const unsigned sbase =
        (unsigned)(uintptr_t)(&lds_stage[0]) + (unsigned)tid * 128u;

    float acc[8][8];
#pragma unroll
    for (int t = 0; t < 8; ++t)
#pragma unroll
        for (int i = 0; i < 8; ++i) acc[t][i] = 0.0f;

    async_stage_x(sbase, xrow);
    async_stage_x(sbase + STAGE_BYTES, xrow + BLK);

    for (int kb = 0; kb < KB; ++kb) {
        const int cur = kb & 1;

        if (kb + 1 < KB) asm volatile("s_wait_asynccnt 0x8" ::: "memory");
        else             asm volatile("s_wait_asynccnt 0x0" ::: "memory");

        const float4* sp =
            (const float4*)(lds_stage + cur * STAGE_BYTES + tid * 128);
        float4 f[8];
#pragma unroll
        for (int j = 0; j < 8; ++j) f[j] = sp[j];

        float amax = 0.0f;
#pragma unroll
        for (int j = 0; j < 8; ++j) {
            amax = fmaxf(amax, fabsf(f[j].x));
            amax = fmaxf(amax, fabsf(f[j].y));
            amax = fmaxf(amax, fabsf(f[j].z));
            amax = fmaxf(amax, fabsf(f[j].w));
        }
        amax = fmaxf(amax, __shfl_xor(amax, 1, 32));
        amax = fmaxf(amax, __shfl_xor(amax, 2, 32));
        amax = fmaxf(amax, 1e-8f);

        const float scale = amax * (1.0f / 448.0f);
        const float inv   = 448.0f / amax;

        int* dst = (int*)(lds_xq + qrow * XQ_STRIDE + qseg * 32);
#pragma unroll
        for (int j = 0; j < 8; ++j) {
            int d = __builtin_amdgcn_cvt_pk_fp8_f32(f[j].x * inv, f[j].y * inv, 0, false);
            d     = __builtin_amdgcn_cvt_pk_fp8_f32(f[j].z * inv, f[j].w * inv, d, true);
            dst[j] = d;
        }
        if (qseg == 0) lds_as[qrow] = scale;

        __syncthreads();

        if (kb + 2 < KB)
            async_stage_x(sbase + cur * STAGE_BYTES, xrow + (kb + 2) * BLK);

        v16i A;
        {
            const unsigned char* ap =
                lds_xq + (wave_m * 16 + l16) * XQ_STRIDE + half * 16;
#pragma unroll
            for (int c = 0; c < 8; ++c) {
                int2 dd = *(const int2*)(ap + (c >> 1) * 32 + (c & 1) * 8);
                A[2 * c]     = dd.x;
                A[2 * c + 1] = dd.y;
            }
        }

        const float wsc = wscale[wave_n * KB + kb];
        float comb[8];
#pragma unroll
        for (int i = 0; i < 8; ++i)
            comb[i] = lds_as[wave_m * 16 + (half ? i + 8 : i)] * wsc;

        const unsigned char* wbase =
            w + (size_t)(wave_n * 128) * K_DIM + kb * BLK;

#pragma unroll
        for (int nt = 0; nt < 8; ++nt) {
            v16i B;
            const unsigned char* bp =
                wbase + (size_t)(nt * 16 + l16) * K_DIM + half * 16;
#pragma unroll
            for (int q = 0; q < 4; ++q) {
                int4 dd = *(const int4*)(bp + q * 32);
                B[4 * q]     = dd.x;
                B[4 * q + 1] = dd.y;
                B[4 * q + 2] = dd.z;
                B[4 * q + 3] = dd.w;
            }

            v8f zero = {0.f, 0.f, 0.f, 0.f, 0.f, 0.f, 0.f, 0.f};
            v8f p = __builtin_amdgcn_wmma_f32_16x16x128_fp8_fp8(
                A, B, (short)0, zero, false, false);

#pragma unroll
            for (int i = 0; i < 8; ++i)
                acc[nt][i] = fmaf(p[i], comb[i], acc[nt][i]);
        }

        __syncthreads();
    }

#pragma unroll
    for (int nt = 0; nt < 8; ++nt) {
        const int col = wave_n * 128 + nt * 16 + l16;
#pragma unroll
        for (int i = 0; i < 8; ++i) {
            const int r = row0 + wave_m * 16 + (half ? i + 8 : i);
            out[(size_t)r * N_DIM + col] = acc[nt][i];
        }
    }
}

extern "C" void kernel_launch(void* const* d_in, const int* in_sizes, int n_in,
                              void* d_out, int out_size, void* d_ws, size_t ws_size,
                              hipStream_t stream) {
    const float*         x  = (const float*)d_in[0];
    const unsigned char* w  = (const unsigned char*)d_in[1];
    const float*         ws = (const float*)d_in[2];
    float*               y  = (float*)d_out;

    const int M = in_sizes[0] / K_DIM;
    dim3 grid(M / ROWS_WG);
    dim3 block(256);
    fp8_gemm_dsv3_kernel<<<grid, block, 0, stream>>>(x, w, ws, y, M);
}
